// mLSTMBlockStack_43739946943132
// MI455X (gfx1250) — compile-verified, matrix operand order checked by reading, not run on hardware
//
#include <hip/hip_runtime.h>

// ---------------- problem constants ----------------
#define SLEN  4000
#define EMB   256
#define INNER 512
#define NHD   2
#define DH    256
#define KCONV 4
#define NB    2

typedef __attribute__((ext_vector_type(16))) __bf16 bf16x16;
typedef __attribute__((ext_vector_type(8)))  float  f32x8;

union FragU { bf16x16 v; uint4 q[2]; };

__device__ __forceinline__ unsigned short f2bf(float f) {
  union { float f; unsigned u; } x; x.f = f;
  unsigned r = x.u + 0x7FFFu + ((x.u >> 16) & 1u);
  return (unsigned short)(r >> 16);
}

// Load a 16x32 bf16 WMMA fragment (A layout; also used for B where the
// "column-owner" row is contiguous in K). rowptr points at the lane's row,
// chunks are at +half*8 and +16+half*8 (16B each, 16B aligned).
__device__ __forceinline__ bf16x16 ld_frag(const unsigned short* rowptr, int half) {
  FragU u;
  u.q[0] = *(const uint4*)(rowptr + half * 8);
  u.q[1] = *(const uint4*)(rowptr + 16 + half * 8);
  return u.v;
}

__device__ __forceinline__ f32x8 wmma_bf16(bf16x16 a, bf16x16 b, f32x8 c) {
  return __builtin_amdgcn_wmma_f32_16x16x32_bf16(false, a, false, b, (short)0, c, false, false);
}

// CDNA5 async copy: 16B per lane, global -> LDS, tracked by ASYNCcnt.
__device__ __forceinline__ void async_ld16(const unsigned short* g, unsigned lds_off) {
  asm volatile("global_load_async_to_lds_b128 %0, %1, off"
               :: "v"(lds_off), "v"((unsigned long long)(size_t)g)
               : "memory");
}

__device__ __forceinline__ float block_reduce_sum(float v, float* sm) {
  int tid = threadIdx.x;
  sm[tid] = v; __syncthreads();
  for (int s = blockDim.x >> 1; s > 0; s >>= 1) {
    if (tid < s) sm[tid] += sm[tid + s];
    __syncthreads();
  }
  float r = sm[0]; __syncthreads();
  return r;
}

__device__ __forceinline__ float logsig(float x) {
  return (x >= 0.f) ? -log1pf(__expf(-x)) : (x - log1pf(__expf(x)));
}

// ---------------- small kernels ----------------
__global__ void cvt_bf16_kernel(const float* __restrict__ in, unsigned short* __restrict__ out, int n) {
  int i = blockIdx.x * blockDim.x + threadIdx.x;
  if (i < n) out[i] = f2bf(in[i]);
}

// LayerNorm over EMB=256, one row per block (256 threads).
__global__ void ln_kernel(const float* __restrict__ x, const float* __restrict__ w,
                          unsigned short* __restrict__ out_bf, float* __restrict__ out_f) {
  __shared__ float sm[256];
  int s = blockIdx.x, tid = threadIdx.x;
  float v = x[s * EMB + tid];
  float mean = block_reduce_sum(v, sm) * (1.f / EMB);
  float d = v - mean;
  float var = block_reduce_sum(d * d, sm) * (1.f / EMB);
  float y = d * rsqrtf(var + 1e-5f) * w[tid];
  if (out_bf) out_bf[s * EMB + tid] = f2bf(y);
  if (out_f)  out_f[s * EMB + tid] = y;
}

// up = xln @ pU^T : (S x 256) x (256 x 1024). grid (S/16, 4), block 128 (4 waves).
__global__ void __launch_bounds__(128) gemm_up_kernel(const unsigned short* __restrict__ xln,
                                                      const unsigned short* __restrict__ pUb,
                                                      float* __restrict__ up) {
  int lane = threadIdx.x & 31, wave = threadIdx.x >> 5;
  int half = lane >> 4, nn = lane & 15;
  int m0 = blockIdx.x * 16;
  int n0 = blockIdx.y * 256 + wave * 64;
  f32x8 acc[4];
  #pragma unroll
  for (int j = 0; j < 4; j++) { f32x8 z = {}; acc[j] = z; }
  const unsigned short* arow = xln + (m0 + nn) * EMB;
  for (int kc = 0; kc < EMB / 32; ++kc) {
    bf16x16 a = ld_frag(arow + kc * 32, half);
    #pragma unroll
    for (int j = 0; j < 4; j++) {
      bf16x16 b = ld_frag(pUb + (n0 + j * 16 + nn) * EMB + kc * 32, half);
      acc[j] = wmma_bf16(a, b, acc[j]);
    }
  }
  #pragma unroll
  for (int j = 0; j < 4; j++)
    #pragma unroll
    for (int i = 0; i < 8; i++)
      up[(m0 + i + 8 * half) * (2 * INNER) + n0 + j * 16 + nn] = acc[j][i];
}

// depthwise causal conv (K=4) + SiLU on x_m = up[:, :INNER]
__global__ void conv_silu_kernel(const float* __restrict__ up, const float* __restrict__ cw,
                                 const float* __restrict__ cb, float* __restrict__ xc) {
  int idx = blockIdx.x * blockDim.x + threadIdx.x;
  if (idx >= SLEN * INNER) return;
  int s = idx / INNER, c = idx % INNER;
  float acc = cb[c];
  #pragma unroll
  for (int j = 0; j < KCONV; j++) {
    int sp = s - (KCONV - 1) + j;
    if (sp >= 0) acc += up[sp * (2 * INNER) + c] * cw[c * KCONV + j];
  }
  xc[idx] = acc / (1.f + __expf(-acc));
}

// headwise 4x4 projections: q,k from xc; v from x_m. Also emits bf16 q/k and vT.
__global__ void headwise_kernel(const float* __restrict__ xc, const float* __restrict__ up,
                                const float* __restrict__ qW, const float* __restrict__ kW,
                                const float* __restrict__ vW,
                                float* __restrict__ qf, float* __restrict__ kf, float* __restrict__ vf,
                                unsigned short* __restrict__ qb2, unsigned short* __restrict__ kb2,
                                unsigned short* __restrict__ vtb) {
  int idx = blockIdx.x * blockDim.x + threadIdx.x;
  if (idx >= SLEN * INNER) return;
  int s = idx / INNER, c = idx % INNER;
  int hh = c >> 2, e = c & 3;
  const float* xr = xc + s * INNER + hh * 4;
  const float* xm = up + s * (2 * INNER) + hh * 4;
  float q = 0.f, k = 0.f, v = 0.f;
  #pragma unroll
  for (int d = 0; d < 4; d++) {
    q += xr[d] * qW[(hh * 4 + e) * 4 + d];
    k += xr[d] * kW[(hh * 4 + e) * 4 + d];
    v += xm[d] * vW[(hh * 4 + e) * 4 + d];
  }
  qf[idx] = q; kf[idx] = k; vf[idx] = v;
  qb2[idx] = f2bf(q); kb2[idx] = f2bf(k);
  vtb[c * SLEN + s] = f2bf(v);   // transposed per head: channel-major
}

// gate GEMVs: ig/fg[h][s] = dot(concat(q,k,v)[s], W[h]) + b[h]
__global__ void gates_kernel(const float* __restrict__ qf, const float* __restrict__ kf,
                             const float* __restrict__ vf,
                             const float* __restrict__ iW, const float* __restrict__ ib,
                             const float* __restrict__ fW, const float* __restrict__ fb,
                             float* __restrict__ ig, float* __restrict__ fg) {
  __shared__ float sm[4][128];
  int s = blockIdx.x, tid = threadIdx.x;
  float a0 = 0, a1 = 0, a2 = 0, a3 = 0;
  for (int i = tid; i < 3 * INNER; i += 128) {
    float xv = (i < INNER) ? qf[s * INNER + i]
             : (i < 2 * INNER) ? kf[s * INNER + i - INNER]
             : vf[s * INNER + i - 2 * INNER];
    a0 += xv * iW[i];  a1 += xv * iW[3 * INNER + i];
    a2 += xv * fW[i];  a3 += xv * fW[3 * INNER + i];
  }
  sm[0][tid] = a0; sm[1][tid] = a1; sm[2][tid] = a2; sm[3][tid] = a3;
  __syncthreads();
  for (int st = 64; st > 0; st >>= 1) {
    if (tid < st)
      for (int r = 0; r < 4; r++) sm[r][tid] += sm[r][tid + st];
    __syncthreads();
  }
  if (tid == 0) {
    ig[0 * SLEN + s] = sm[0][0] + ib[0];
    ig[1 * SLEN + s] = sm[1][0] + ib[1];
    fg[0 * SLEN + s] = sm[2][0] + fb[0];
    fg[1 * SLEN + s] = sm[3][0] + fb[1];
  }
}

// per-head inclusive scan of logsigmoid(fg): CS[h][0]=0, CS[h][i+1]=sum_{j<=i} lf[j]
__global__ void scan_kernel(const float* __restrict__ fg, float* __restrict__ CS) {
  __shared__ float sm[256];
  __shared__ float carry;
  int h = blockIdx.x, tid = threadIdx.x;
  if (tid == 0) { carry = 0.f; CS[h * (SLEN + 1)] = 0.f; }
  __syncthreads();
  for (int base = 0; base < SLEN; base += 256) {
    int i = base + tid;
    sm[tid] = (i < SLEN) ? logsig(fg[h * SLEN + i]) : 0.f;
    __syncthreads();
    for (int off = 1; off < 256; off <<= 1) {
      float t = (tid >= off) ? sm[tid - off] : 0.f;
      __syncthreads();
      sm[tid] += t;
      __syncthreads();
    }
    if (i < SLEN) CS[h * (SLEN + 1) + i + 1] = carry + sm[tid];
    __syncthreads();
    if (tid == 0) carry += sm[255];
    __syncthreads();
  }
}

// ---------------- flash-style mLSTM cell (WMMA + async-LDS pipeline) ----------------
// grid (ceil(S/64), NHD), block 128 = 4 waves, each wave owns 16 query rows.
// K/V tiles for the whole block are staged cooperatively into double-buffered LDS
// with GLOBAL_LOAD_ASYNC_TO_LDS_B128 (ASYNCcnt), one tile ahead of compute.
__global__ void __launch_bounds__(128) attn_kernel(const unsigned short* __restrict__ qb2,
                                                   const unsigned short* __restrict__ kb2,
                                                   const unsigned short* __restrict__ vtb,
                                                   const float* __restrict__ CS,
                                                   const float* __restrict__ IG,
                                                   float* __restrict__ hbuf) {
  int tid = threadIdx.x;
  int lane = tid & 31, wave = tid >> 5;
  int half = lane >> 4, nn = lane & 15;
  int n = blockIdx.y;
  int s0 = blockIdx.x * 64 + wave * 16;
  bool active = (s0 < SLEN);
  int blockSmax = blockIdx.x * 64 + 63;
  if (blockSmax > SLEN - 1) blockSmax = SLEN - 1;

  __shared__ unsigned short Kbuf[2][32 * 256];   // [t-local][ch]   16KB x2
  __shared__ unsigned short Vbuf[2][256 * 32];   // [d][t-local]    16KB x2
  __shared__ unsigned short Plds[4][16 * 32];
  unsigned short* pl = Plds[wave];

  const float* CSn = CS + n * (SLEN + 1);
  const float* IGn = IG + n * SLEN;
  const unsigned short* kbh = kb2 + n * DH;
  const unsigned short* vth = vtb + n * DH * SLEN;

  // cooperative async stage of one 32-key tile into LDS buffer nb
  auto issue_tile = [&](int t0, int nb) {
    unsigned short* Kb = Kbuf[nb];
    unsigned short* Vb = Vbuf[nb];
    #pragma unroll
    for (int i = 0; i < 8; i++) {
      int idx = tid + i * 128;
      int tl = idx >> 5, ck = idx & 31;                 // K: 32 rows x 32 chunks
      async_ld16(kbh + (t0 + tl) * INNER + ck * 8,
                 (unsigned)(size_t)(Kb + tl * 256 + ck * 8));
      int d = idx >> 2, cv = idx & 3;                   // V: 256 rows x 4 chunks
      async_ld16(vth + d * SLEN + t0 + cv * 8,
                 (unsigned)(size_t)(Vb + d * 32 + cv * 8));
    }
  };

  // resident Q fragments (16 rows x DH=256 -> 8 k-chunks)
  bf16x16 aq[8];
  float csr[8], mrow[8], den[8];
  if (active) {
    const unsigned short* qrow = qb2 + (s0 + nn) * INNER + n * DH;
    #pragma unroll
    for (int kc = 0; kc < 8; kc++) aq[kc] = ld_frag(qrow + kc * 32, half);
    #pragma unroll
    for (int i = 0; i < 8; i++) {
      csr[i] = CSn[s0 + i + 8 * half + 1];
      mrow[i] = -1e30f;
      den[i] = 0.f;
    }
  }
  f32x8 acc[16];
  #pragma unroll
  for (int j = 0; j < 16; j++) { f32x8 z = {}; acc[j] = z; }

  const float scale = 0.0625f;  // DH^-0.5
  const int smax = s0 + 15;

  issue_tile(0, 0);  // prologue

  for (int t0 = 0; t0 <= blockSmax; t0 += 32) {
    int buf = (t0 >> 5) & 1;
    asm volatile("s_wait_asynccnt 0" ::: "memory");  // tile t0 is in LDS
    __syncthreads();
    if (t0 + 32 <= blockSmax) issue_tile(t0 + 32, buf ^ 1);  // overlap next fetch

    if (active && t0 <= smax) {
      const unsigned short* Kb = Kbuf[buf];
      const unsigned short* Vb = Vbuf[buf];

      // S = Q K^T over the 32-key tile (two 16-wide column halves)
      f32x8 sc[2];
      float cst[2], igt[2];
      #pragma unroll
      for (int h2 = 0; h2 < 2; h2++) {
        int t = t0 + h2 * 16 + nn;
        const unsigned short* krow = Kb + (h2 * 16 + nn) * 256;
        f32x8 c = {};
        #pragma unroll
        for (int kc = 0; kc < 8; kc++) {
          bf16x16 b = ld_frag(krow + kc * 32, half);
          c = wmma_bf16(aq[kc], b, c);
        }
        sc[h2] = c;
        cst[h2] = CSn[t + 1];
        igt[h2] = IGn[t];
      }

      // decay logits + causal mask + tile rowmax
      float logd[2][8], rm[8];
      #pragma unroll
      for (int i = 0; i < 8; i++) {
        int srow = s0 + i + 8 * half;
        float lm = -1e30f;
        #pragma unroll
        for (int h2 = 0; h2 < 2; h2++) {
          int t = t0 + h2 * 16 + nn;
          float ld = (t <= srow) ? (csr[i] - cst[h2] + igt[h2]) : -1e30f;
          logd[h2][i] = ld;
          lm = fmaxf(lm, ld);
        }
        rm[i] = lm;
      }
      #pragma unroll
      for (int m = 1; m < 16; m <<= 1)
        #pragma unroll
        for (int i = 0; i < 8; i++) rm[i] = fmaxf(rm[i], __shfl_xor(rm[i], m, 32));

      // flash rescale
      float corr[8];
      #pragma unroll
      for (int i = 0; i < 8; i++) {
        float newm = fmaxf(mrow[i], rm[i]);
        corr[i] = (newm <= -1e29f) ? 0.f : __expf(mrow[i] - newm);
        mrow[i] = newm;
      }
      float rs[8];
      #pragma unroll
      for (int i = 0; i < 8; i++) {
        float p0 = (logd[0][i] <= -1e29f) ? 0.f : sc[0][i] * scale * __expf(logd[0][i] - mrow[i]);
        float p1 = (logd[1][i] <= -1e29f) ? 0.f : sc[1][i] * scale * __expf(logd[1][i] - mrow[i]);
        sc[0][i] = p0; sc[1][i] = p1;
        rs[i] = p0 + p1;
      }
      #pragma unroll
      for (int m = 1; m < 16; m <<= 1)
        #pragma unroll
        for (int i = 0; i < 8; i++) rs[i] += __shfl_xor(rs[i], m, 32);
      #pragma unroll
      for (int i = 0; i < 8; i++) den[i] = den[i] * corr[i] + rs[i];
      #pragma unroll
      for (int j = 0; j < 16; j++)
        #pragma unroll
        for (int i = 0; i < 8; i++) acc[j][i] *= corr[i];

      // stage P (C-layout) -> LDS bf16 -> reload as A-layout fragment
      #pragma unroll
      for (int h2 = 0; h2 < 2; h2++)
        #pragma unroll
        for (int i = 0; i < 8; i++)
          pl[(i + 8 * half) * 32 + h2 * 16 + nn] = f2bf(sc[h2][i]);
      asm volatile("s_wait_dscnt 0" ::: "memory");
      bf16x16 ap = ld_frag(pl + nn * 32, half);

      // num += P @ V  (16 output column tiles over DH)
      #pragma unroll
      for (int j = 0; j < 16; j++) {
        bf16x16 b = ld_frag(Vb + (j * 16 + nn) * 32, half);
        acc[j] = wmma_bf16(ap, b, acc[j]);
      }
    }
  }

  if (active) {
    float inv[8];
    #pragma unroll
    for (int i = 0; i < 8; i++)
      inv[i] = 1.f / (fmaxf(fabsf(den[i]), __expf(-mrow[i])) + 1e-6f);
    #pragma unroll
    for (int j = 0; j < 16; j++)
      #pragma unroll
      for (int i = 0; i < 8; i++)
        hbuf[(s0 + i + 8 * half) * INNER + n * DH + j * 16 + nn] = acc[j][i] * inv[i];
  }
}

// per-head LN of h, + skip*xc, * silu(z) -> yb (bf16)
__global__ void mix_kernel(const float* __restrict__ hbuf, const float* __restrict__ xc,
                           const float* __restrict__ up, const float* __restrict__ onw,
                           const float* __restrict__ skw, unsigned short* __restrict__ yb) {
  __shared__ float sm[256];
  int s = blockIdx.x, tid = threadIdx.x;
  for (int h = 0; h < NHD; h++) {
    int c = h * DH + tid;
    float v = hbuf[s * INNER + c];
    float mean = block_reduce_sum(v, sm) * (1.f / DH);
    float d = v - mean;
    float var = block_reduce_sum(d * d, sm) * (1.f / DH);
    float hn = d * rsqrtf(var + 1e-5f) * onw[c];
    float hs = hn + skw[c] * xc[s * INNER + c];
    float z = up[s * (2 * INNER) + INNER + c];
    float y = hs * (z / (1.f + __expf(-z)));
    yb[s * INNER + c] = f2bf(y);
  }
}

// x += y @ pD^T : (S x 512) x (512 x 256). grid S/16, block 128 (4 waves cover N=256).
__global__ void __launch_bounds__(128) gemm_down_kernel(const unsigned short* __restrict__ yb,
                                                        const unsigned short* __restrict__ pDb,
                                                        float* __restrict__ xio) {
  int lane = threadIdx.x & 31, wave = threadIdx.x >> 5;
  int half = lane >> 4, nn = lane & 15;
  int m0 = blockIdx.x * 16;
  int n0 = wave * 64;
  f32x8 acc[4];
  #pragma unroll
  for (int j = 0; j < 4; j++) { f32x8 z = {}; acc[j] = z; }
  const unsigned short* arow = yb + (m0 + nn) * INNER;
  for (int kc = 0; kc < INNER / 32; ++kc) {
    bf16x16 a = ld_frag(arow + kc * 32, half);
    #pragma unroll
    for (int j = 0; j < 4; j++) {
      bf16x16 b = ld_frag(pDb + (n0 + j * 16 + nn) * INNER + kc * 32, half);
      acc[j] = wmma_bf16(a, b, acc[j]);
    }
  }
  #pragma unroll
  for (int j = 0; j < 4; j++)
    #pragma unroll
    for (int i = 0; i < 8; i++) {
      int m = m0 + i + 8 * half, cc = n0 + j * 16 + nn;
      xio[m * EMB + cc] += acc[j][i];
    }
}

// ---------------- host orchestration ----------------
extern "C" void kernel_launch(void* const* d_in, const int* in_sizes, int n_in,
                              void* d_out, int out_size, void* d_ws, size_t ws_size,
                              hipStream_t stream) {
  (void)in_sizes; (void)n_in; (void)out_size; (void)ws_size;
  const float* x_in  = (const float*)d_in[0];
  const float* ln_w  = (const float*)d_in[1];
  const float* pU    = (const float*)d_in[2];
  const float* convW = (const float*)d_in[3];
  const float* convb = (const float*)d_in[4];
  const float* qW    = (const float*)d_in[5];
  const float* kW    = (const float*)d_in[6];
  const float* vW    = (const float*)d_in[7];
  const float* igW   = (const float*)d_in[8];
  const float* igb   = (const float*)d_in[9];
  const float* fgW   = (const float*)d_in[10];
  const float* fgb   = (const float*)d_in[11];
  const float* onw   = (const float*)d_in[12];
  const float* skw   = (const float*)d_in[13];
  const float* pD    = (const float*)d_in[14];
  const float* pnw   = (const float*)d_in[15];

  char* wsp = (char*)d_ws;
  size_t off = 0;
  auto alloc = [&](size_t bytes) -> void* {
    void* p = wsp + off;
    off = (off + bytes + 255) & ~(size_t)255;
    return p;
  };
  float*          xbuf = (float*)alloc((size_t)SLEN * EMB * 4);
  unsigned short* xlnb = (unsigned short*)alloc((size_t)SLEN * EMB * 2);
  float*          up   = (float*)alloc((size_t)SLEN * 2 * INNER * 4);
  float*          xc   = (float*)alloc((size_t)SLEN * INNER * 4);
  float*          qf   = (float*)alloc((size_t)SLEN * INNER * 4);
  float*          kf   = (float*)alloc((size_t)SLEN * INNER * 4);
  float*          vf   = (float*)alloc((size_t)SLEN * INNER * 4);
  unsigned short* qb2  = (unsigned short*)alloc((size_t)SLEN * INNER * 2);
  unsigned short* kb2  = (unsigned short*)alloc((size_t)SLEN * INNER * 2);
  unsigned short* vtb  = (unsigned short*)alloc((size_t)NHD * DH * SLEN * 2);
  float*          ig   = (float*)alloc((size_t)NHD * SLEN * 4);
  float*          fg   = (float*)alloc((size_t)NHD * SLEN * 4);
  float*          CS   = (float*)alloc((size_t)NHD * (SLEN + 1) * 4);
  float*          hbuf = (float*)alloc((size_t)SLEN * INNER * 4);
  unsigned short* yb   = (unsigned short*)alloc((size_t)SLEN * INNER * 2);
  unsigned short* pUb  = (unsigned short*)alloc((size_t)2 * INNER * EMB * 2);
  unsigned short* pDb  = (unsigned short*)alloc((size_t)EMB * INNER * 2);

  hipMemcpyAsync(xbuf, x_in, (size_t)SLEN * EMB * 4, hipMemcpyDeviceToDevice, stream);

  const int elems = SLEN * INNER;
  for (int b = 0; b < NB; b++) {
    cvt_bf16_kernel<<<(2 * INNER * EMB + 255) / 256, 256, 0, stream>>>(pU + (size_t)b * 2 * INNER * EMB, pUb, 2 * INNER * EMB);
    cvt_bf16_kernel<<<(EMB * INNER + 255) / 256, 256, 0, stream>>>(pD + (size_t)b * EMB * INNER, pDb, EMB * INNER);

    ln_kernel<<<SLEN, 256, 0, stream>>>(xbuf, ln_w + b * EMB, xlnb, nullptr);
    gemm_up_kernel<<<dim3(SLEN / 16, (2 * INNER) / 256), 128, 0, stream>>>(xlnb, pUb, up);
    conv_silu_kernel<<<(elems + 255) / 256, 256, 0, stream>>>(up, convW + b * INNER * KCONV, convb + b * INNER, xc);
    headwise_kernel<<<(elems + 255) / 256, 256, 0, stream>>>(xc, up,
        qW + b * 128 * 16, kW + b * 128 * 16, vW + b * 128 * 16,
        qf, kf, vf, qb2, kb2, vtb);
    gates_kernel<<<SLEN, 128, 0, stream>>>(qf, kf, vf,
        igW + b * NHD * 3 * INNER, igb + b * NHD,
        fgW + b * NHD * 3 * INNER, fgb + b * NHD, ig, fg);
    scan_kernel<<<NHD, 256, 0, stream>>>(fg, CS);
    attn_kernel<<<dim3((SLEN + 63) / 64, NHD), 128, 0, stream>>>(qb2, kb2, vtb, CS, ig, hbuf);
    mix_kernel<<<SLEN, 256, 0, stream>>>(hbuf, xc, up, onw + b * INNER, skw + b * INNER, yb);
    gemm_down_kernel<<<SLEN / 16, 128, 0, stream>>>(yb, pDb, xbuf);
  }
  ln_kernel<<<SLEN, 256, 0, stream>>>(xbuf, pnw, nullptr, (float*)d_out);
}
